// SO2MultiheadAttention_86208583566068
// MI455X (gfx1250) — compile-verified, matrix operand order checked by reading, not run on hardware
//
#include <hip/hip_runtime.h>
#include <hip/hip_bf16.h>
#include <math.h>
#include <stdint.h>

// ---- problem constants (from reference) ----
#define BBATCH 4
#define SEQ    1024
#define HEADS  8
#define DMODEL 3584
#define DHEAD  448     // DMODEL / HEADS

typedef __bf16 bf16_t;
typedef __attribute__((ext_vector_type(16))) __bf16 bf16x16;
typedef __attribute__((ext_vector_type(8)))  __bf16 bf16x8;
typedef __attribute__((ext_vector_type(8)))  float  f32x8;

// ---------------------------------------------------------------------------
// async global->LDS copy of 16 bytes per lane (CDNA5 path, ASYNCcnt-tracked)
// lds addr = low 32 bits of generic pointer (aperture maps addr[31:0] -> LDS)
// ---------------------------------------------------------------------------
__device__ __forceinline__ void async_cp16(uint32_t ldsAddr, const bf16_t* g) {
  asm volatile("global_load_async_to_lds_b128 %0, %1, off"
               :: "v"(ldsAddr), "v"(g) : "memory");
}
__device__ __forceinline__ void wait_async0() {
  asm volatile("s_wait_asynccnt 0" ::: "memory");
}

// ---------------------------------------------------------------------------
// fp32 -> bf16 elementwise convert (vectorized x4)
// ---------------------------------------------------------------------------
__global__ __launch_bounds__(256)
void cvt_f32_to_bf16(const float* __restrict__ in, bf16_t* __restrict__ out, int n) {
  int i = (blockIdx.x * blockDim.x + threadIdx.x) * 4;
  if (i + 3 < n) {
    float4 v = *(const float4*)(in + i);
    out[i + 0] = (bf16_t)v.x;
    out[i + 1] = (bf16_t)v.y;
    out[i + 2] = (bf16_t)v.z;
    out[i + 3] = (bf16_t)v.w;
  }
}

// ---------------------------------------------------------------------------
// W (D,D) fp32 row-major -> Wt (D,D) bf16 with Wt[n][k] = W[k][n]
// ---------------------------------------------------------------------------
__global__ __launch_bounds__(256)
void transpose_w_to_bf16(const float* __restrict__ W, bf16_t* __restrict__ Wt, int Dd) {
  __shared__ float tile[32][33];
  int n0 = blockIdx.x * 32;
  int k0 = blockIdx.y * 32;
  int tx = threadIdx.x, ty = threadIdx.y;
#pragma unroll
  for (int j = 0; j < 4; ++j)
    tile[ty + j * 8][tx] = W[(size_t)(k0 + ty + j * 8) * Dd + n0 + tx];
  __syncthreads();
#pragma unroll
  for (int j = 0; j < 4; ++j)
    Wt[(size_t)(n0 + ty + j * 8) * Dd + k0 + tx] = (bf16_t)tile[tx][ty + j * 8];
}

// ---------------------------------------------------------------------------
// v (B,S,D) bf16 -> vt (B*H, DHEAD, SEQ) bf16   (PV GEMM becomes K-contiguous)
// ---------------------------------------------------------------------------
__global__ __launch_bounds__(256)
void transpose_v(const bf16_t* __restrict__ v, bf16_t* __restrict__ vt) {
  __shared__ bf16_t tile[32][33];
  int z  = blockIdx.z;           // b*H + h
  int b  = z / HEADS;
  int h  = z % HEADS;
  int d0 = blockIdx.x * 32;      // dh
  int s0 = blockIdx.y * 32;      // s
  int tx = threadIdx.x, ty = threadIdx.y;
#pragma unroll
  for (int j = 0; j < 4; ++j)
    tile[ty + j * 8][tx] =
        v[((size_t)b * SEQ + s0 + ty + j * 8) * DMODEL + h * DHEAD + d0 + tx];
  __syncthreads();
#pragma unroll
  for (int j = 0; j < 4; ++j)
    vt[((size_t)z * DHEAD + d0 + ty + j * 8) * SEQ + s0 + tx] = tile[tx][ty + j * 8];
}

// ---------------------------------------------------------------------------
// masked softmax over rows of 1024 (logits fp32 -> attn bf16)
// ---------------------------------------------------------------------------
__global__ __launch_bounds__(256)
void softmax_mask(const float* __restrict__ logits, const int* __restrict__ mask,
                  bf16_t* __restrict__ attn, float scale) {
  __shared__ float red[256];
  int r  = blockIdx.x;               // (b*H + h)*SEQ + q
  int q  = r & (SEQ - 1);
  int bh = r >> 10;
  int b  = bh >> 3;
  const float* L  = logits + (size_t)r * SEQ;
  const int*   Mk = mask + ((size_t)b * SEQ + q) * SEQ;
  int t = threadIdx.x;
  float v[4];
  float mx = -3.0e38f;
#pragma unroll
  for (int i = 0; i < 4; ++i) {
    int idx = t + i * 256;
    float f = L[idx] * scale;
    if (Mk[idx] == 0) f = -9.0e15f;
    v[i] = f;
    mx = fmaxf(mx, f);
  }
  red[t] = mx; __syncthreads();
  for (int s = 128; s > 0; s >>= 1) {
    if (t < s) red[t] = fmaxf(red[t], red[t + s]);
    __syncthreads();
  }
  mx = red[0]; __syncthreads();
  float sum = 0.f;
#pragma unroll
  for (int i = 0; i < 4; ++i) { float e = __expf(v[i] - mx); v[i] = e; sum += e; }
  red[t] = sum; __syncthreads();
  for (int s = 128; s > 0; s >>= 1) {
    if (t < s) red[t] += red[t + s];
    __syncthreads();
  }
  float inv = 1.f / red[0];
  bf16_t* Ar = attn + (size_t)r * SEQ;
#pragma unroll
  for (int i = 0; i < 4; ++i) {
    int idx = t + i * 256;
    Ar[idx] = (bf16_t)(v[i] * inv);
  }
}

// ---------------------------------------------------------------------------
// Generic batched bf16 GEMM via v_wmma_f32_16x16x32_bf16
//   C[m][n] (+batch base) = sum_k A[m][k] * Bt[n][k]
// Block tile 256x128, BK=32, 256 threads = 8 waves (4x2), each wave 64x64
// (4x4 WMMA 16x16 tiles). Double-buffered LDS filled by async-to-LDS copies.
// Batch base for X = (z/div)*xOuter + (z%div)*xInner.  REQUIRES M % 256 == 0.
// ---------------------------------------------------------------------------
#define TP 40   // LDS tile pitch in elems (80 B: 16B-aligned, conflict-free)
#define BM 256
#define BN 128
#define BK 32

__device__ __forceinline__ bf16x16 load_frag(const bf16_t* p, int half) {
  // per-lane fragment: elems 0..7 = K half*8..+7, elems 8..15 = K 16+half*8..+7
  bf16x8 lo = *(const bf16x8*)(p + half * 8);
  bf16x8 hi = *(const bf16x8*)(p + 16 + half * 8);
  bf16x16 r;
#pragma unroll
  for (int i = 0; i < 8; ++i) { r[i] = lo[i]; r[i + 8] = hi[i]; }
  return r;
}

__global__ __launch_bounds__(256)
void gemm_bf16_wmma(const bf16_t* __restrict__ A, long long aOuter, long long aInner, int lda,
                    const bf16_t* __restrict__ Bt, long long bOuter, long long bInner, int ldb,
                    void* __restrict__ Cv, long long cOuter, long long cInner, int ldc,
                    int M, int N, int K, int divi, int outBf16) {
  __shared__ bf16_t sA[2][BM * TP];
  __shared__ bf16_t sB[2][BN * TP];

  int z  = blockIdx.z;
  int zo = z / divi, zi = z % divi;
  const bf16_t* Ab = A  + (size_t)(zo * aOuter + zi * aInner);
  const bf16_t* Bb = Bt + (size_t)(zo * bOuter + zi * bInner);

  int m0 = blockIdx.y * BM;
  int n0 = blockIdx.x * BN;
  int tid  = threadIdx.x;
  int lane = tid & 31;
  int wave = tid >> 5;
  int wm = (wave >> 1) * 64;   // wave m-offset (4 rows of waves)
  int wn = (wave & 1) * 64;    // wave n-offset (2 cols of waves)
  int half = lane >> 4;
  int frow = lane & 15;

  // per-thread staging coordinates (16B per async copy)
  int colS  = (tid * 8) & 31;
  int rowS0 = (tid * 8) >> 5;    // +64 per chunk

  f32x8 acc[4][4];
#pragma unroll
  for (int mi = 0; mi < 4; ++mi)
#pragma unroll
    for (int ni = 0; ni < 4; ++ni)
#pragma unroll
      for (int e = 0; e < 8; ++e) acc[mi][ni][e] = 0.f;

  auto stage = [&](int buf, int k0) {
    const bf16_t* pa = Ab + (size_t)(m0 + rowS0) * lda + k0 + colS;
    uint32_t la = (uint32_t)(uintptr_t)&sA[buf][rowS0 * TP + colS];
#pragma unroll
    for (int i = 0; i < 4; ++i)
      async_cp16(la + (uint32_t)(i * 64 * TP * 2), pa + (size_t)i * 64 * lda);
#pragma unroll
    for (int i = 0; i < 2; ++i) {
      int gn = n0 + rowS0 + i * 64;
      if (gn >= N) gn = N - 1;           // N-tail: clamp (cols >= N never stored)
      const bf16_t* pb = Bb + (size_t)gn * ldb + k0 + colS;
      uint32_t lb = (uint32_t)(uintptr_t)&sB[buf][(rowS0 + i * 64) * TP + colS];
      async_cp16(lb, pb);
    }
  };

  int nk = K / BK;
  stage(0, 0);
  wait_async0();
  __syncthreads();

  for (int t = 0; t < nk; ++t) {
    int cur = t & 1;
    if (t + 1 < nk) stage(cur ^ 1, (t + 1) * BK);   // overlap with compute

    const bf16_t* tA = sA[cur];
    const bf16_t* tB = sB[cur];
    bf16x16 aF[4], bF[4];
#pragma unroll
    for (int mi = 0; mi < 4; ++mi)
      aF[mi] = load_frag(tA + (wm + mi * 16 + frow) * TP, half);
#pragma unroll
    for (int ni = 0; ni < 4; ++ni)
      bF[ni] = load_frag(tB + (wn + ni * 16 + frow) * TP, half);

#pragma unroll
    for (int mi = 0; mi < 4; ++mi)
#pragma unroll
      for (int ni = 0; ni < 4; ++ni)
        acc[mi][ni] = __builtin_amdgcn_wmma_f32_16x16x32_bf16(
            false, aF[mi], false, bF[ni], (short)0, acc[mi][ni], false, false);

    wait_async0();       // own async writes of next buffer landed
    __syncthreads();     // everyone done reading cur + writing next
  }

  // epilogue: C/D layout — VGPR e: m = e + (lane>=16)*8, n = lane&15
  long long cBase = (long long)zo * cOuter + (long long)zi * cInner;
  int gmBase = m0 + wm + (lane >> 4) * 8;
  int gnBase = n0 + wn + (lane & 15);
  if (outBf16) {
    bf16_t* C = (bf16_t*)Cv + cBase;
#pragma unroll
    for (int mi = 0; mi < 4; ++mi) {
      bf16_t* rowP = C + (size_t)(gmBase + mi * 16) * ldc;
#pragma unroll
      for (int ni = 0; ni < 4; ++ni) {
        int gn = gnBase + ni * 16;
        if (gn < N) {
          bf16_t* p = rowP + gn;
#pragma unroll
          for (int e = 0; e < 8; ++e) { *p = (bf16_t)acc[mi][ni][e]; p += ldc; }
        }
      }
    }
  } else {
    float* C = (float*)Cv + cBase;
#pragma unroll
    for (int mi = 0; mi < 4; ++mi) {
      float* rowP = C + (size_t)(gmBase + mi * 16) * ldc;
#pragma unroll
      for (int ni = 0; ni < 4; ++ni) {
        int gn = gnBase + ni * 16;
        if (gn < N) {
          float* p = rowP + gn;
#pragma unroll
          for (int e = 0; e < 8; ++e) { *p = acc[mi][ni][e]; p += ldc; }
        }
      }
    }
  }
}

// ---------------------------------------------------------------------------
// host-side launch
// ---------------------------------------------------------------------------
extern "C" void kernel_launch(void* const* d_in, const int* in_sizes, int n_in,
                              void* d_out, int out_size, void* d_ws, size_t ws_size,
                              hipStream_t stream) {
  (void)in_sizes; (void)n_in; (void)out_size; (void)ws_size;

  const float* x    = (const float*)d_in[0];
  const float* Wq   = (const float*)d_in[1];
  const float* Wk   = (const float*)d_in[2];
  const float* Wv   = (const float*)d_in[3];
  const float* Wo   = (const float*)d_in[4];
  const int*   mask = (const int*)d_in[5];
  float*       out  = (float*)d_out;

  const size_t XB_E  = (size_t)BBATCH * SEQ * DMODEL;      // 14,680,064
  const size_t WB_E  = (size_t)DMODEL * DMODEL;            // 12,845,056
  const size_t LOG_E = (size_t)BBATCH * HEADS * SEQ * SEQ; // 33,554,432

  size_t off = 0;
  auto wsAlloc = [&](size_t bytes) -> void* {
    void* p = (void*)((char*)d_ws + off);
    off += (bytes + 255) & ~(size_t)255;
    return p;
  };

  bf16_t* xb   = (bf16_t*)wsAlloc(XB_E * 2);
  bf16_t* Wtq  = (bf16_t*)wsAlloc(WB_E * 2);
  bf16_t* Wtk  = (bf16_t*)wsAlloc(WB_E * 2);
  bf16_t* Wtv  = (bf16_t*)wsAlloc(WB_E * 2);
  bf16_t* Wto  = (bf16_t*)wsAlloc(WB_E * 2);
  bf16_t* qb   = (bf16_t*)wsAlloc(XB_E * 2);
  bf16_t* kb   = (bf16_t*)wsAlloc(XB_E * 2);
  bf16_t* vb   = (bf16_t*)wsAlloc(XB_E * 2);
  bf16_t* vt   = (bf16_t*)wsAlloc(XB_E * 2);
  float*  lgt  = (float*)wsAlloc(LOG_E * 4);
  bf16_t* attn = (bf16_t*)wsAlloc(LOG_E * 2);
  bf16_t* vals = (bf16_t*)wsAlloc(XB_E * 2);

  // 1) x -> bf16
  cvt_f32_to_bf16<<<dim3((unsigned)(XB_E / 4 / 256)), dim3(256), 0, stream>>>(x, xb, (int)XB_E);

  // 2) weights -> transposed bf16
  dim3 gW(DMODEL / 32, DMODEL / 32), bW(32, 8);
  transpose_w_to_bf16<<<gW, bW, 0, stream>>>(Wq, Wtq, DMODEL);
  transpose_w_to_bf16<<<gW, bW, 0, stream>>>(Wk, Wtk, DMODEL);
  transpose_w_to_bf16<<<gW, bW, 0, stream>>>(Wv, Wtv, DMODEL);
  transpose_w_to_bf16<<<gW, bW, 0, stream>>>(Wo, Wto, DMODEL);

  const long long SD = (long long)SEQ * DMODEL;
  // 3) Q/K/V projections: (4096x3584) = xb @ Wt^T
  dim3 gProj(DMODEL / BN, (BBATCH * SEQ) / BM, 1);     // (28, 16, 1)
  gemm_bf16_wmma<<<gProj, 256, 0, stream>>>(xb, 0, 0, DMODEL, Wtq, 0, 0, DMODEL,
                                            qb, 0, 0, DMODEL,
                                            BBATCH * SEQ, DMODEL, DMODEL, 1, 1);
  gemm_bf16_wmma<<<gProj, 256, 0, stream>>>(xb, 0, 0, DMODEL, Wtk, 0, 0, DMODEL,
                                            kb, 0, 0, DMODEL,
                                            BBATCH * SEQ, DMODEL, DMODEL, 1, 1);
  gemm_bf16_wmma<<<gProj, 256, 0, stream>>>(xb, 0, 0, DMODEL, Wtv, 0, 0, DMODEL,
                                            vb, 0, 0, DMODEL,
                                            BBATCH * SEQ, DMODEL, DMODEL, 1, 1);

  // 4) v -> (B*H, DHEAD, SEQ)
  transpose_v<<<dim3(DHEAD / 32, SEQ / 32, BBATCH * HEADS), dim3(32, 8), 0, stream>>>(vb, vt);

  // 5) logits = q @ k^T per (b,h); q/k read in-place from (B,S,D) layout
  dim3 gLog(SEQ / BN, SEQ / BM, BBATCH * HEADS);       // (8, 4, 32)
  gemm_bf16_wmma<<<gLog, 256, 0, stream>>>(
      qb, SD, DHEAD, DMODEL,
      kb, SD, DHEAD, DMODEL,
      lgt, (long long)HEADS * SEQ * SEQ, (long long)SEQ * SEQ, SEQ,
      SEQ, SEQ, DHEAD, HEADS, 0);

  // 6) masked softmax (scale 1/sqrt(DH)), fp32 -> bf16
  softmax_mask<<<dim3(BBATCH * HEADS * SEQ), dim3(256), 0, stream>>>(
      lgt, mask, attn, 1.0f / sqrtf((float)DHEAD));

  // 7) vals = attn @ v  (per (b,h); N=448 tail handled by guards)
  dim3 gAV((DHEAD + BN - 1) / BN, SEQ / BM, BBATCH * HEADS);  // (4, 4, 32)
  gemm_bf16_wmma<<<gAV, 256, 0, stream>>>(
      attn, (long long)HEADS * SEQ * SEQ, (long long)SEQ * SEQ, SEQ,
      vt, (long long)HEADS * DHEAD * SEQ, (long long)DHEAD * SEQ, SEQ,
      vals, SD, DHEAD, DMODEL,
      SEQ, DHEAD, SEQ, HEADS, 1);

  // 8) out = vals @ Wo  (fp32 output)
  gemm_bf16_wmma<<<gProj, 256, 0, stream>>>(vals, 0, 0, DMODEL, Wto, 0, 0, DMODEL,
                                            out, 0, 0, DMODEL,
                                            BBATCH * SEQ, DMODEL, DMODEL, 1, 0);
}
